// BevModel_38955353375395
// MI455X (gfx1250) — matrix operand order repaired and checked by reading, not compiled, not run on hardware
//
#include <hip/hip_runtime.h>
#include <math.h>

// ---------------- types for WMMA (CDNA5 / gfx1250, wave32) ----------------
typedef __bf16 bf16_t;
typedef __attribute__((ext_vector_type(16))) __bf16 v16bf;
typedef __attribute__((ext_vector_type(8)))  __bf16 v8bf;
typedef __attribute__((ext_vector_type(8)))  float  v8f;

// ---------------- problem constants (match reference) ----------------
#define BATCH 2
#define NCAM  6
#define NIMG  12          // BATCH*NCAM
#define IMH   256
#define IMW   704
#define HD    32
#define WD    88
#define NPIX  2816        // HD*WD
#define MTOT  33792       // NIMG*NPIX (GEMM rows)
#define KTOT  192         // 3*8*8 patch size
#define NF    64          // feat channels
#define NDEP  48          // depth bins
#define NTOT  112         // NF + NDEP
#define XDIM  200
#define YDIM  200
#define NVOX  40000
#define NPTS  1622016     // BATCH*NCAM*NDEP*NPIX
#define BEV_ELEMS 5120000 // BATCH*NVOX*NF

#define AP 200            // padded LDS row stride (bf16) for A tiles
#define WP 200            // padded LDS row stride (bf16) for weights

__device__ __forceinline__ bf16_t f2bf(float f) {
    union { float f; unsigned u; } v; v.f = f;
    unsigned r = v.u + 0x7FFFu + ((v.u >> 16) & 1u);   // round-to-nearest-even
    unsigned short h = (unsigned short)(r >> 16);
    bf16_t out;
    __builtin_memcpy(&out, &h, 2);
    return out;
}

// ------------------------------------------------------------------
// Kernel 1: per-camera matrices. comb = inv(E)[:3,:3] @ inv(K), trans.
// E is a rigid transform -> closed-form inverse; K via adjugate.
// mats[i*12 + 0..8] = comb row-major, [9..11] = trans
// ------------------------------------------------------------------
__global__ void precompute_mats(const float* __restrict__ intr,
                                const float* __restrict__ extr,
                                float* __restrict__ mats) {
    int i = threadIdx.x;
    if (i >= NIMG) return;
    const float* E = extr + i * 16;
    const float* K = intr + i * 9;
    float R[3][3], tI[3];
    #pragma unroll
    for (int r = 0; r < 3; r++)
        #pragma unroll
        for (int c = 0; c < 3; c++)
            R[r][c] = E[c * 4 + r];                       // transpose of rotation block
    #pragma unroll
    for (int r = 0; r < 3; r++)
        tI[r] = -(R[r][0] * E[3] + R[r][1] * E[7] + R[r][2] * E[11]);

    float a = K[0], b = K[1], c = K[2], d = K[3], e = K[4], f = K[5],
          g = K[6], h = K[7], ii = K[8];
    float A0 =  (e * ii - f * h);
    float B0 = -(d * ii - f * g);
    float C0 =  (d * h  - e * g);
    float inv = 1.0f / (a * A0 + b * B0 + c * C0);
    float Ki[3][3];
    Ki[0][0] = A0 * inv;              Ki[0][1] = -(b * ii - c * h) * inv; Ki[0][2] =  (b * f - c * e) * inv;
    Ki[1][0] = B0 * inv;              Ki[1][1] =  (a * ii - c * g) * inv; Ki[1][2] = -(a * f - c * d) * inv;
    Ki[2][0] = C0 * inv;              Ki[2][1] = -(a * h  - b * g) * inv; Ki[2][2] =  (a * e - b * d) * inv;

    float* M = mats + i * 12;
    #pragma unroll
    for (int r = 0; r < 3; r++)
        #pragma unroll
        for (int cc = 0; cc < 3; cc++)
            M[r * 3 + cc] = R[r][0] * Ki[0][cc] + R[r][1] * Ki[1][cc] + R[r][2] * Ki[2][cc];
    M[9] = tI[0]; M[10] = tI[1]; M[11] = tI[2];
}

// ------------------------------------------------------------------
// Kernel 2: zero fill
// ------------------------------------------------------------------
__global__ void zero_f32(float* __restrict__ p, int n) {
    int i = blockIdx.x * blockDim.x + threadIdx.x;
    if (i < n) p[i] = 0.0f;
}

// ------------------------------------------------------------------
// Kernel 3: fused patchify GEMM via WMMA bf16 -> f32.
//   A: [MTOT x 192] patches gathered from images
//   B: [192 x 112]  (cols 0..63 = w_feat, 64..111 = w_depth)
// Block = 64 threads = 2 waves; each wave owns one 16-row M tile and
// the full N=112 (7 x 16 tiles); K = 6 steps of 32.
// ------------------------------------------------------------------
__global__ __launch_bounds__(64)
void patch_gemm(const float* __restrict__ images,
                const float* __restrict__ w_feat,  const float* __restrict__ b_feat,
                const float* __restrict__ w_depth, const float* __restrict__ b_depth,
                float* __restrict__ feat_ws, float* __restrict__ dep_ws) {
    __shared__ __align__(16) bf16_t wl[NTOT * WP];   // weights [n][k], padded
    __shared__ __align__(16) bf16_t al[2 * 16 * AP]; // A tiles per wave [m][k], padded

    const int tid  = threadIdx.x;
    const int wave = tid >> 5;
    const int lane = tid & 31;
    const int tile0 = blockIdx.x * 2;

    // ---- stage weights (bf16) ----
    for (int idx = tid; idx < NTOT * KTOT; idx += 64) {
        int n = idx / KTOT, k = idx - n * KTOT;
        float v = (n < NF) ? w_feat[n * KTOT + k] : w_depth[(n - NF) * KTOT + k];
        wl[n * WP + k] = f2bf(v);
    }
    // ---- stage A tiles: gather 8x8x3 patches ----
    for (int idx = tid; idx < 2 * 16 * KTOT; idx += 64) {
        int wv  = idx / (16 * KTOT);
        int rem = idx - wv * 16 * KTOT;
        int r = rem / KTOT, k = rem - r * KTOT;
        int m   = (tile0 + wv) * 16 + r;
        int img = m / NPIX, pix = m - img * NPIX;
        int oh  = pix / WD, ow = pix - oh * WD;
        int ic  = k >> 6, kh = (k >> 3) & 7, kw = k & 7;
        float v = images[((img * 3 + ic) * IMH + oh * 8 + kh) * IMW + ow * 8 + kw];
        al[(wv * 16 + r) * AP + k] = f2bf(v);
    }
    __syncthreads();

    const int half = lane >> 4;       // 0 or 1
    const int mn   = lane & 15;       // m row (A) / n col (B,D)
    v8f acc[7] = {};
    const bf16_t* aw = al + wave * 16 * AP;

    #pragma unroll
    for (int ks = 0; ks < 6; ks++) {
        // A fragment: e<8 -> K=ks*32+8*half+e ; e>=8 -> +16
        const bf16_t* ap = aw + mn * AP + ks * 32 + half * 8;
        v8bf alo = *(const v8bf*)(ap);
        v8bf ahi = *(const v8bf*)(ap + 16);
        v16bf afrag;
        #pragma unroll
        for (int i = 0; i < 8; i++) { afrag[i] = alo[i]; afrag[8 + i] = ahi[i]; }

        #pragma unroll
        for (int t = 0; t < 7; t++) {
            // B fragment: e<8 -> K=ks*32+8*half+e ; e>=8 -> +16, at column n = t*16+mn
            const bf16_t* bp = wl + (t * 16 + mn) * WP + ks * 32 + half * 8;
            v8bf blo = *(const v8bf*)(bp);
            v8bf bhi = *(const v8bf*)(bp + 16);
            v16bf bfrag;
            #pragma unroll
            for (int i = 0; i < 8; i++) { bfrag[i] = blo[i]; bfrag[8 + i] = bhi[i]; }

            acc[t] = __builtin_amdgcn_wmma_f32_16x16x32_bf16(
                false, afrag, false, bfrag, (short)0, acc[t], false, false);
        }
    }

    // ---- writeback: D element r -> m = r + 8*half, n = t*16+mn ----
    const int mbase = (tile0 + wave) * 16;
    #pragma unroll
    for (int t = 0; t < 7; t++) {
        int n = t * 16 + mn;
        float bias = (n < NF) ? b_feat[n] : b_depth[n - NF];
        #pragma unroll
        for (int r = 0; r < 8; r++) {
            int m = mbase + r + 8 * half;
            float v = acc[t][r] + bias;
            if (n < NF) feat_ws[m * NF + n] = v;
            else        dep_ws[m * NDEP + (n - NF)] = v;
        }
    }
}

// ------------------------------------------------------------------
// Kernel 4: softmax over the 48 depth bins; writes depth_prob output
// in reference layout (NIMG, D, HD*WD).
// ------------------------------------------------------------------
__global__ void depth_softmax(const float* __restrict__ dep_ws,
                              float* __restrict__ outp) {
    int t = blockIdx.x * blockDim.x + threadIdx.x;
    if (t >= MTOT) return;
    const float* row = dep_ws + t * NDEP;
    float mx = row[0];
    #pragma unroll
    for (int d = 1; d < NDEP; d++) mx = fmaxf(mx, row[d]);
    float e[NDEP];
    float s = 0.0f;
    #pragma unroll
    for (int d = 0; d < NDEP; d++) { e[d] = __expf(row[d] - mx); s += e[d]; }
    float inv = 1.0f / s;
    int img = t / NPIX, pix = t - img * NPIX;
    float* o = outp + img * (NDEP * NPIX) + pix;
    #pragma unroll
    for (int d = 0; d < NDEP; d++) o[d * NPIX] = e[d] * inv;
}

// ------------------------------------------------------------------
// Kernel 5: fused outer-product + BEV splat.
// 4 points per 256-thread block; 64 threads (= 2 waves) per point.
// Accumulator layout [b][voxel][c] so each point's 64 atomics are
// 256 contiguous bytes (stays in L2).
// ------------------------------------------------------------------
__global__ __launch_bounds__(256)
void bev_scatter(const float* __restrict__ mats,
                 const float* __restrict__ feat_ws,
                 const float* __restrict__ depth_prob,
                 float* __restrict__ bev_ws) {
    __shared__ int   sVox[4];
    __shared__ int   sFeat[4];
    __shared__ float sDp[4];
    int tid = threadIdx.x;
    if (tid < 4) {
        int P   = blockIdx.x * 4 + tid;            // NPTS = 405504*4 exactly
        int b   = P / (NCAM * NDEP * NPIX);
        int rem = P - b * (NCAM * NDEP * NPIX);
        int n   = rem / (NDEP * NPIX);  rem -= n * (NDEP * NPIX);
        int d   = rem / NPIX;
        int pix = rem - d * NPIX;
        int oh  = pix / WD, ow = pix - oh * WD;
        int img = b * NCAM + n;
        float dep = 2.0f + (float)d;               // D0 + d*DSTEP
        float px = (float)ow * (703.0f / 87.0f) * dep;   // linspace(0,W-1,WD)
        float py = (float)oh * (255.0f / 31.0f) * dep;   // linspace(0,H-1,HD)
        float pz = dep;
        const float* M = mats + img * 12;
        float gx = M[0] * px + M[1] * py + M[2] * pz + M[9];
        float gy = M[3] * px + M[4] * py + M[5] * pz + M[10];
        float gz = M[6] * px + M[7] * py + M[8] * pz + M[11];
        // (geom - (START - RES/2)) / RES, truncated toward zero like astype(int32)
        int ix = (int)((gx + 50.0f) * 2.0f);
        int iy = (int)((gy + 50.0f) * 2.0f);
        int iz = (int)((gz + 10.0f) * 0.05f);
        bool ok = (ix >= 0) && (ix < XDIM) && (iy >= 0) && (iy < YDIM) && (iz == 0);
        sVox[tid]  = ok ? ((b * NVOX + ix * YDIM + iy) * NF) : -1;
        sFeat[tid] = (img * NPIX + pix) * NF;
        sDp[tid]   = depth_prob[img * (NDEP * NPIX) + d * NPIX + pix];
    }
    __syncthreads();
    int p = tid >> 6, c = tid & 63;
    int vb = sVox[p];
    if (vb >= 0) {
        atomicAdd(&bev_ws[vb + c], feat_ws[sFeat[p] + c] * sDp[p]);
    }
}

// ------------------------------------------------------------------
// Kernel 6: transpose [b][voxel][c] -> output (B, C, X, Y) via LDS tile.
// 64 voxels x 64 channels per block; 40000 = 625 * 64 chunks per batch.
// ------------------------------------------------------------------
__global__ __launch_bounds__(256)
void bev_transpose(const float* __restrict__ bev_ws, float* __restrict__ out) {
    __shared__ float tile[64][65];
    int blk   = blockIdx.x;
    int b     = blk / 625;
    int vbase = (blk - b * 625) * 64;
    int tid   = threadIdx.x;
    #pragma unroll
    for (int i = 0; i < 16; i++) {
        int idx = tid + i * 256;
        int vl = idx >> 6, c = idx & 63;
        tile[c][vl] = bev_ws[(b * NVOX + vbase + vl) * NF + c];
    }
    __syncthreads();
    #pragma unroll
    for (int i = 0; i < 16; i++) {
        int idx = tid + i * 256;
        int c = idx >> 6, vl = idx & 63;
        out[(b * NF + c) * NVOX + vbase + vl] = tile[c][vl];
    }
}

// ------------------------------------------------------------------
extern "C" void kernel_launch(void* const* d_in, const int* in_sizes, int n_in,
                              void* d_out, int out_size, void* d_ws, size_t ws_size,
                              hipStream_t stream) {
    const float* images  = (const float*)d_in[0];
    const float* intr    = (const float*)d_in[1];
    const float* extr    = (const float*)d_in[2];
    const float* w_feat  = (const float*)d_in[3];
    const float* b_feat  = (const float*)d_in[4];
    const float* w_depth = (const float*)d_in[5];
    const float* b_depth = (const float*)d_in[6];

    float* out      = (float*)d_out;
    float* outDepth = out + (size_t)BATCH * NF * NVOX;   // +5,120,000

    char*  ws      = (char*)d_ws;
    float* mats    = (float*)ws;                         // 144 floats
    float* bev_ws  = (float*)(ws + 1024);                // 5,120,000 floats
    float* feat_ws = bev_ws + BEV_ELEMS;                 // 2,162,688 floats
    float* dep_ws  = feat_ws + (size_t)MTOT * NF;        // 1,622,016 floats

    precompute_mats<<<1, 32, 0, stream>>>(intr, extr, mats);
    zero_f32<<<(BEV_ELEMS + 255) / 256, 256, 0, stream>>>(bev_ws, BEV_ELEMS);
    patch_gemm<<<MTOT / 32, 64, 0, stream>>>(images, w_feat, b_feat,
                                             w_depth, b_depth, feat_ws, dep_ws);
    depth_softmax<<<(MTOT + 127) / 128, 128, 0, stream>>>(dep_ws, outDepth);
    bev_scatter<<<NPTS / 4, 256, 0, stream>>>(mats, feat_ws, outDepth, bev_ws);
    bev_transpose<<<BATCH * 625, 256, 0, stream>>>(bev_ws, out);
}
